// Entropic_Sinkhorn_Loss_86311662780619
// MI455X (gfx1250) — matrix operand order repaired and checked by reading, not compiled, not run on hardware
//
#include <hip/hip_runtime.h>
#include <hip/hip_bf16.h>

#define NN 8192
#define DD 256
#define RBLK 64          // rows per block in K-sweep kernels
#define NRB (NN / RBLK)  // 128 row-blocks

typedef __attribute__((ext_vector_type(16))) _Float16 v16h;
typedef __attribute__((ext_vector_type(8)))  _Float16 v8h;
typedef __attribute__((ext_vector_type(8)))  float    v8f;

// ---------------- f32 -> f16 conversion (once; inputs reused 512x in GEMM) --
__global__ void cvt_f16_kernel(const float* __restrict__ in,
                               _Float16* __restrict__ out, int count) {
    int i = blockIdx.x * blockDim.x + threadIdx.x;
    if (i < count) out[i] = (_Float16)in[i];
}

// ---------------- fragment loaders ----------------------------------------
// A fragment (ISA 7.12.2, 16-bit A 16x32): lane L holds row L%16,
// halves 0..7 = K ka..ka+7, halves 8..15 = K ka+16..ka+23, ka=(L>=16?8:0).
__device__ __forceinline__ v16h load_afrag(const _Float16* __restrict__ A,
                                           int arow, int k0, int ka) {
    v8h a0 = *(const v8h*)(A + (size_t)arow * DD + k0 + ka);
    v8h a1 = *(const v8h*)(A + (size_t)arow * DD + k0 + ka + 16);
    return __builtin_shufflevector(a0, a1,
        0, 1, 2, 3, 4, 5, 6, 7, 8, 9, 10, 11, 12, 13, 14, 15);
}
// B fragment (32x16): lane L holds col L%16, halves 0..7 = K kb..kb+7,
// halves 8..15 = K kb+16..kb+23, kb = (L>=16 ? 8 : 0).
__device__ __forceinline__ v16h load_bfrag(const _Float16* __restrict__ B,
                                           int bcol, int k0, int kb) {
    v8h b0 = *(const v8h*)(B + (size_t)bcol * DD + k0 + kb);
    v8h b1 = *(const v8h*)(B + (size_t)bcol * DD + k0 + kb + 16);
    return __builtin_shufflevector(b0, b1,
        0, 1, 2, 3, 4, 5, 6, 7, 8, 9, 10, 11, 12, 13, 14, 15);
}

// ---------------- GEMM (img @ txt^T) + K = exp((S-1)*100), fp32 out --------
// One wave computes a 16x64 strip (4 WMMA tiles), K-loop of 8 x (16x16x32),
// software-pipelined one K-step ahead so WMMAs overlap next step's loads.
__global__ __launch_bounds__(256)
void gemm_exp_kernel(const _Float16* __restrict__ A,   // imgH [NN][DD]
                     const _Float16* __restrict__ B,   // txtH [NN][DD]
                     float* __restrict__ K) {
    const int lane = threadIdx.x & 31;
    const int wave = threadIdx.x >> 5;
    const int m0 = blockIdx.y * 16;
    const int n0 = (blockIdx.x * 8 + wave) * 64;

    const int arow = m0 + (lane & 15);
    const int ka = (lane >> 4) * 8;
    const int kb = (lane >> 4) * 8;
    const int bc = (lane & 15);

    v8f acc[4] = {};

    v16h a_cur = load_afrag(A, arow, 0, ka);
    v16h b_cur[4];
#pragma unroll
    for (int t = 0; t < 4; ++t)
        b_cur[t] = load_bfrag(B, n0 + t * 16 + bc, 0, kb);

#pragma unroll
    for (int k0 = 0; k0 < DD; k0 += 32) {
        v16h a_nxt = {};
        v16h b_nxt[4] = {};
        if (k0 + 32 < DD) {
            a_nxt = load_afrag(A, arow, k0 + 32, ka);
#pragma unroll
            for (int t = 0; t < 4; ++t)
                b_nxt[t] = load_bfrag(B, n0 + t * 16 + bc, k0 + 32, kb);
        }
#pragma unroll
        for (int t = 0; t < 4; ++t)
            acc[t] = __builtin_amdgcn_wmma_f32_16x16x32_f16(
                false, a_cur, false, b_cur[t], (short)0, acc[t], false, false);
        a_cur = a_nxt;
#pragma unroll
        for (int t = 0; t < 4; ++t) b_cur[t] = b_nxt[t];
    }

    // C/D layout: VGPR e, lanes 0-15 -> M=e, lanes 16-31 -> M=e+8, N=lane%16
    const int rbase = m0 + ((lane >> 4) << 3);
#pragma unroll
    for (int t = 0; t < 4; ++t) {
        const int col = n0 + t * 16 + bc;
#pragma unroll
        for (int e = 0; e < 8; ++e) {
            float s = acc[t][e];
            K[(size_t)(rbase + e) * NN + col] = __expf((s - 1.0f) * 100.0f);
        }
    }
}

// ---------------- small helpers -------------------------------------------
__global__ void zero_kernel(float* p, int count) {
    int i = blockIdx.x * blockDim.x + threadIdx.x;
    if (i < count) p[i] = 0.0f;
}

__global__ void initvec_kernel(float* u1, float* v1, float* u2, float* v2,
                               float val) {
    int i = blockIdx.x * blockDim.x + threadIdx.x;
    if (i < NN) { u1[i] = val; v1[i] = val; u2[i] = val; v2[i] = val; }
}

__global__ void recip_kernel(const float* __restrict__ in1, float* __restrict__ out1,
                             const float* __restrict__ in2, float* __restrict__ out2,
                             float numer) {
    int i = blockIdx.x * blockDim.x + threadIdx.x;
    if (i < NN) { out1[i] = numer / in1[i]; out2[i] = numer / in2[i]; }
}

// ---------------- fused dual matvec: one sweep of K gives K*x and K^T*y ----
// Block = 256 threads, tile = RBLK rows x 256 cols (fully coalesced).
// outRow[i] += sum_j K[i][j]*wRow[j]; outCol[j] += sum_i K[i][j]*wCol[i]
__global__ __launch_bounds__(256)
void dual_matvec_kernel(const float* __restrict__ K,
                        const float* __restrict__ wRow,
                        const float* __restrict__ wCol,
                        float* __restrict__ outRow,
                        float* __restrict__ outCol) {
    __shared__ float wc[RBLK];
    __shared__ float rowAcc[RBLK];
    const int tid  = threadIdx.x;
    const int lane = tid & 31;
    const int c  = blockIdx.x * 256 + tid;
    const int r0 = blockIdx.y * RBLK;

    if (tid < RBLK) { wc[tid] = wCol[r0 + tid]; rowAcc[tid] = 0.0f; }
    __syncthreads();

    const float wr = wRow[c];
    float colAcc = 0.0f;
    for (int r = 0; r < RBLK; ++r) {
        if (r + 1 < RBLK)
            __builtin_prefetch(K + (size_t)(r0 + r + 1) * NN + c, 0, 3);
        float val = K[(size_t)(r0 + r) * NN + c];
        colAcc += val * wc[r];
        float rc = val * wr;
#pragma unroll
        for (int off = 16; off > 0; off >>= 1)
            rc += __shfl_down(rc, off, 32);
        if (lane == 0) atomicAdd(&rowAcc[r], rc);  // ds_add_f32
    }
    __syncthreads();
    if (tid < RBLK) atomicAdd(&outRow[r0 + tid], rowAcc[tid]);
    atomicAdd(&outCol[c], colAcc);
}

// ---------------- P1 loss: streaming softmax over K rows (coalesced) -------
// P1[i][j] = u[i]*K[i][j]*v[j]; label = i -> diagonal term.
__global__ __launch_bounds__(256)
void row_loss_kernel(const float* __restrict__ K,
                     const float* __restrict__ u, const float* __restrict__ v,
                     float scale, float* __restrict__ out) {
    const int i    = blockIdx.x;
    const int tid  = threadIdx.x;
    const int lane = tid & 31;
    const int wave = tid >> 5;
    const float ui = u[i];

    float m = -3.0e38f, s = 0.0f;
    for (int j = tid; j < NN; j += 256) {
        float x = ui * K[(size_t)i * NN + j] * v[j];
        if (x > m) { s = s * __expf(m - x) + 1.0f; m = x; }
        else       { s += __expf(x - m); }
    }
#pragma unroll
    for (int off = 16; off > 0; off >>= 1) {
        float om = __shfl_xor(m, off, 32);
        float os = __shfl_xor(s, off, 32);
        float m2 = fmaxf(m, om);
        s = s * __expf(m - m2) + os * __expf(om - m2);
        m = m2;
    }
    __shared__ float sm[8], ss[8];
    if (lane == 0) { sm[wave] = m; ss[wave] = s; }
    __syncthreads();
    if (tid == 0) {
        float M = sm[0], S = ss[0];
        for (int w = 1; w < 8; ++w) {
            float m2 = fmaxf(M, sm[w]);
            S = S * __expf(M - m2) + ss[w] * __expf(sm[w] - m2);
            M = m2;
        }
        float xii = ui * K[(size_t)i * NN + i] * v[i];
        atomicAdd(out, -(xii - M - __logf(S)) * scale);
    }
}

// ---------------- P2 loss stage 1: coalesced per-column softmax partials ---
// P2[i][j] = u2[i]*K[j][i]*v2[j] -> row i of P2 is column i of K.
// Each thread owns K-column c, streams RBLK rows (coalesced), emits (m,s).
__global__ __launch_bounds__(256)
void col_loss_partial_kernel(const float* __restrict__ K,
                             const float* __restrict__ u2,
                             const float* __restrict__ v2,
                             float* __restrict__ pm, float* __restrict__ ps) {
    __shared__ float vr[RBLK];
    const int tid = threadIdx.x;
    const int c  = blockIdx.x * 256 + tid;
    const int r0 = blockIdx.y * RBLK;

    if (tid < RBLK) vr[tid] = v2[r0 + tid];
    __syncthreads();

    const float uc = u2[c];
    float m = -3.0e38f, s = 0.0f;
    for (int r = 0; r < RBLK; ++r) {
        if (r + 1 < RBLK)
            __builtin_prefetch(K + (size_t)(r0 + r + 1) * NN + c, 0, 3);
        float x = uc * K[(size_t)(r0 + r) * NN + c] * vr[r];
        if (x > m) { s = s * __expf(m - x) + 1.0f; m = x; }
        else       { s += __expf(x - m); }
    }
    pm[(size_t)blockIdx.y * NN + c] = m;
    ps[(size_t)blockIdx.y * NN + c] = s;
}

// ---------------- P2 loss stage 2: fold partials, add diagonal CE ----------
__global__ __launch_bounds__(256)
void col_loss_combine_kernel(const float* __restrict__ K,
                             const float* __restrict__ u2,
                             const float* __restrict__ v2,
                             const float* __restrict__ pm,
                             const float* __restrict__ ps,
                             float scale, float* __restrict__ out) {
    const int c = blockIdx.x * 256 + threadIdx.x;
    float M = -3.0e38f, S = 0.0f;
    for (int rb = 0; rb < NRB; ++rb) {
        float om = pm[(size_t)rb * NN + c];
        float os = ps[(size_t)rb * NN + c];
        float m2 = fmaxf(M, om);
        S = S * __expf(M - m2) + os * __expf(om - m2);
        M = m2;
    }
    float xii = u2[c] * K[(size_t)c * NN + c] * v2[c];
    atomicAdd(out, -(xii - M - __logf(S)) * scale);
}

// ---------------- driver ----------------------------------------------------
extern "C" void kernel_launch(void* const* d_in, const int* in_sizes, int n_in,
                              void* d_out, int out_size, void* d_ws, size_t ws_size,
                              hipStream_t stream) {
    (void)in_sizes; (void)n_in; (void)out_size; (void)ws_size;
    const float* img = (const float*)d_in[0];
    const float* txt = (const float*)d_in[1];
    float* out = (float*)d_out;

    // workspace layout
    float* Kbuf = (float*)d_ws;                       // NN*NN fp32 (256 MiB)
    size_t nn = (size_t)NN * NN;
    float* u1 = Kbuf + nn;
    float* v1 = u1 + NN;
    float* u2 = v1 + NN;
    float* v2 = u2 + NN;
    float* rRow = v2 + NN;                            // row-space accum
    float* rCol = rRow + NN;                          // col-space accum (contiguous)
    _Float16* imgH = (_Float16*)(rCol + NN);
    _Float16* txtH = imgH + (size_t)NN * DD;
    float* pm = (float*)(txtH + (size_t)NN * DD);     // NRB*NN partial maxes (4 MB)
    float* ps = pm + (size_t)NRB * NN;                // NRB*NN partial sums  (4 MB)

    const int nd = NN * DD;
    cvt_f16_kernel<<<(nd + 255) / 256, 256, 0, stream>>>(img, imgH, nd);
    cvt_f16_kernel<<<(nd + 255) / 256, 256, 0, stream>>>(txt, txtH, nd);

    // GEMM + exp: grid (8192/(8*64)=16, 8192/16=512), 8 waves/block
    dim3 gg(NN / (8 * 64), NN / 16);
    gemm_exp_kernel<<<gg, 256, 0, stream>>>(imgH, txtH, Kbuf);

    const float b = 1.0f / NN;
    initvec_kernel<<<(NN + 255) / 256, 256, 0, stream>>>(u1, v1, u2, v2, b);

    dim3 mg(NN / 256, NRB);
    for (int it = 0; it < 5; ++it) {
        // pass 1: rCol = K^T u1, rRow = K u2
        zero_kernel<<<(2 * NN + 255) / 256, 256, 0, stream>>>(rRow, 2 * NN);
        dual_matvec_kernel<<<mg, 256, 0, stream>>>(Kbuf, u2, u1, rRow, rCol);
        // v1 = b / (K^T u1), v2 = b / (K u2)
        recip_kernel<<<(NN + 255) / 256, 256, 0, stream>>>(rCol, v1, rRow, v2, b);
        // pass 2: rRow = K v1, rCol = K^T v2
        zero_kernel<<<(2 * NN + 255) / 256, 256, 0, stream>>>(rRow, 2 * NN);
        dual_matvec_kernel<<<mg, 256, 0, stream>>>(Kbuf, v1, v2, rRow, rCol);
        // u1 = 1 / (K v1), u2 = 1 / (K^T v2)
        recip_kernel<<<(NN + 255) / 256, 256, 0, stream>>>(rRow, u1, rCol, u2, 1.0f);
    }

    // loss = ( mean row-CE(P1) + mean row-CE(P2) ) / 2
    zero_kernel<<<1, 256, 0, stream>>>(out, 1);
    const float scale = 1.0f / (2.0f * NN);
    row_loss_kernel<<<NN, 256, 0, stream>>>(Kbuf, u1, v1, scale, out);
    col_loss_partial_kernel<<<mg, 256, 0, stream>>>(Kbuf, u2, v2, pm, ps);
    col_loss_combine_kernel<<<NN / 256, 256, 0, stream>>>(Kbuf, u2, v2, pm, ps,
                                                          scale, out);
}
